// GraphSAGE_57071525429588
// MI455X (gfx1250) — compile-verified, matrix operand order checked by reading, not run on hardware
//
#include <hip/hip_runtime.h>

#define NODES 50000
#define EDGES 800000
#define DIM   128
#define NCLS  40

typedef __attribute__((ext_vector_type(16))) _Float16 v16h;
typedef __attribute__((ext_vector_type(8)))  _Float16 v8h;
typedef __attribute__((ext_vector_type(2)))  _Float16 h2;
typedef __attribute__((ext_vector_type(8)))  float    v8f;
typedef __attribute__((ext_vector_type(4)))  int      i4;

typedef __attribute__((address_space(1))) i4 gi4;   // global int4
typedef __attribute__((address_space(3))) i4 si4;   // LDS int4

// ---- gfx1250 async global->LDS copy (16B), guarded fallback to sync copy ----
__device__ __forceinline__ void async_copy16(const _Float16* g, _Float16* l) {
#if __has_builtin(__builtin_amdgcn_global_load_async_to_lds_b128)
    __builtin_amdgcn_global_load_async_to_lds_b128((gi4*)g, (si4*)l, 0, 0);
#else
    *(float4*)l = *(const float4*)g;   // synchronous fallback (vmem load + ds store)
#endif
}
__device__ __forceinline__ void async_wait_all() {
#if __has_builtin(__builtin_amdgcn_s_wait_asynccnt)
    __builtin_amdgcn_s_wait_asynccnt(0);
#else
    asm volatile("s_wait_asynccnt 0x0" ::: "memory");
#endif
}

// ---------------- utility kernels ----------------

__global__ void k_zero(float* __restrict__ p, int n) {
    int t = blockIdx.x * blockDim.x + threadIdx.x;
    if (t < n) p[t] = 0.0f;
}

__global__ void k_f32_to_f16(const float* __restrict__ s, _Float16* __restrict__ d, int n) {
    int t = blockIdx.x * blockDim.x + threadIdx.x;
    if (t < n) d[t] = (_Float16)s[t];
}

__global__ void k_deg(const int* __restrict__ dst, float* __restrict__ deg) {
    int t = blockIdx.x * blockDim.x + threadIdx.x;
    if (t < EDGES) atomicAdd(&deg[dst[t]], 1.0f);
}

__global__ void k_invdeg(const float* __restrict__ deg, float* __restrict__ inv, int n) {
    int t = blockIdx.x * blockDim.x + threadIdx.x;
    if (t < n) inv[t] = 1.0f / fmaxf(deg[t], 1.0f);
}

// ---- weight repack: f32 [K x Dout]x2 -> f16 B-fragment layout ----
// packed[((kt*NT + nt)*32 + lane)*16 + elem] : each lane reads 32 contiguous bytes per K-step.
__global__ void k_pack(const float* __restrict__ wl, const float* __restrict__ wr,
                       _Float16* __restrict__ bp, int K, int Dout, int Nh) {
    int NT = (2 * Nh) >> 4;
    int total = (K >> 5) * NT * 512;
    int t = blockIdx.x * blockDim.x + threadIdx.x;
    if (t >= total) return;
    int elem = t & 15;
    int lane = (t >> 4) & 31;
    int tile = t >> 9;
    int nt = tile % NT, kt = tile / NT;
    int half = lane >> 4;
    int n = nt * 16 + (lane & 15);
    int k = (kt << 5) + ((elem & 8) ? 16 : 0) + half * 8 + (elem & 7);
    const float* w = (n < Nh) ? wl : wr;
    int col = (n < Nh) ? n : n - Nh;
    float v = (col < Dout) ? w[(size_t)k * Dout + col] : 0.0f;
    bp[t] = (_Float16)v;
}

// ---- WMMA GEMM: Y[NODES x N2](f16) = A[NODES x 128](f16) @ Bpacked ----
// Block = 8 waves sharing one 16-row A tile staged in LDS via async copy.
// Each wave computes one 16x16 tile: 4 x v_wmma_f32_16x16x32_f16.
__global__ void k_gemm(const _Float16* __restrict__ A, const _Float16* __restrict__ Bp,
                       _Float16* __restrict__ Y, int NT, int bpm) {
    __shared__ _Float16 tileA[16 * DIM];           // 4 KB
    const int lane = threadIdx.x & 31;
    const int wv   = threadIdx.x >> 5;
    const int mt   = blockIdx.x / bpm;
    const int nt   = (blockIdx.x % bpm) * 8 + wv;

    // cooperative stage: 256 threads x 16B = 16 rows x 128 halves
    {
        int c = threadIdx.x;                        // 0..255
        int row = c >> 4, off = (c & 15) * 8;
        async_copy16(A + ((size_t)(mt * 16 + row)) * DIM + off, tileA + row * DIM + off);
    }
    async_wait_all();
    __syncthreads();

    if (nt < NT) {
        const int half = lane >> 4, mr = lane & 15;
        const int N2 = NT * 16;
        v8f acc = {};
#pragma unroll
        for (int kt = 0; kt < 4; ++kt) {
            const int k0 = kt * 32 + half * 8;      // lane holds K = k0..k0+7, k0+16..k0+23
            v8h lo = *(const v8h*)(tileA + mr * DIM + k0);
            v8h hi = *(const v8h*)(tileA + mr * DIM + k0 + 16);
            v16h a;
#pragma unroll
            for (int i = 0; i < 8; ++i) { a[i] = lo[i]; a[8 + i] = hi[i]; }
            v16h b = *(const v16h*)(Bp + (((size_t)(kt * NT + nt)) * 32 + lane) * 16);
            acc = __builtin_amdgcn_wmma_f32_16x16x32_f16(false, a, false, b,
                                                         (short)0, acc, false, false);
        }
        // C/D layout: VGPR r -> M = r + 8*half, N = lane & 15
        _Float16* yrow = Y + (size_t)(mt * 16 + half * 8) * N2 + nt * 16 + mr;
#pragma unroll
        for (int r = 0; r < 8; ++r) yrow[(size_t)r * N2] = (_Float16)acc[r];
    }
}

// ---- edge scatter-add: agg[dst] += (float)Y_l[src]  (f16 reads, f32 atomic accumulate) ----
__global__ void k_scatter(const int* __restrict__ src, const int* __restrict__ dst,
                          const _Float16* __restrict__ Y, float* __restrict__ agg,
                          int Dout, int N2, int Nh) {
    int t = blockIdx.x * blockDim.x + threadIdx.x;
    int e = t >> 5, lane = t & 31;
    if (e >= EDGES) return;
    int s = src[e], d = dst[e];
    const h2* yr = (const h2*)(Y + (size_t)s * N2);   // y_l part: cols 0..Dout-1
    float* ar = agg + (size_t)d * Nh;
    for (int c = lane; c < (Dout >> 1); c += 32) {
        h2 v = yr[c];
        atomicAdd(&ar[2 * c],     (float)v.x);
        atomicAdd(&ar[2 * c + 1], (float)v.y);
    }
}

// ---- combine: H = relu(agg*invdeg + y_r + b), f16 output for hidden layers ----
__global__ void k_combine16(const float* __restrict__ agg, const _Float16* __restrict__ Y,
                            const float* __restrict__ inv, const float* __restrict__ bias,
                            _Float16* __restrict__ H, int Dout, int N2, int Nh) {
    int t = blockIdx.x * blockDim.x + threadIdx.x;
    if (t >= NODES * Dout) return;
    int n = t / Dout, c = t - n * Dout;
    float v = agg[(size_t)n * Nh + c] * inv[n] + (float)Y[(size_t)n * N2 + Nh + c] + bias[c];
    H[t] = (_Float16)fmaxf(v, 0.0f);
}

__global__ void k_combine32(const float* __restrict__ agg, const _Float16* __restrict__ Y,
                            const float* __restrict__ inv, const float* __restrict__ bias,
                            float* __restrict__ H, int Dout, int N2, int Nh) {
    int t = blockIdx.x * blockDim.x + threadIdx.x;
    if (t >= NODES * Dout) return;
    int n = t / Dout, c = t - n * Dout;
    float v = agg[(size_t)n * Nh + c] * inv[n] + (float)Y[(size_t)n * N2 + Nh + c] + bias[c];
    H[t] = fmaxf(v, 0.0f);
}

// ---------------- host driver ----------------

extern "C" void kernel_launch(void* const* d_in, const int* in_sizes, int n_in,
                              void* d_out, int out_size, void* d_ws, size_t ws_size,
                              hipStream_t stream) {
    const float* x   = (const float*)d_in[0];
    const int*   ei  = (const int*)d_in[1];     // edge_index [2, E]: row0=src, row1=dst
    const int*   src = ei;
    const int*   dst = ei + EDGES;
    const float* w1l = (const float*)d_in[2], *w1r = (const float*)d_in[3], *b1 = (const float*)d_in[4];
    const float* w2l = (const float*)d_in[5], *w2r = (const float*)d_in[6], *b2 = (const float*)d_in[7];
    const float* w3l = (const float*)d_in[8], *w3r = (const float*)d_in[9], *b3 = (const float*)d_in[10];
    float* out = (float*)d_out;

    char* ws = (char*)d_ws;
    size_t off = 0;
    auto carve = [&](size_t bytes) { size_t o = off; off += (bytes + 255) & ~(size_t)255; return o; };
    float*    deg = (float*)(ws + carve((size_t)NODES * 4));
    float*    inv = (float*)(ws + carve((size_t)NODES * 4));
    _Float16* bp  = (_Float16*)(ws + carve((size_t)128 * 256 * 2));
    _Float16* Yh  = (_Float16*)(ws + carve((size_t)NODES * 256 * 2));
    float*    agg = (float*)(ws + carve((size_t)NODES * 128 * 4));
    _Float16* xh  = (_Float16*)(ws + carve((size_t)NODES * DIM * 2));
    _Float16* hA  = (_Float16*)(ws + carve((size_t)NODES * DIM * 2));
    _Float16* hB  = (_Float16*)(ws + carve((size_t)NODES * DIM * 2));

    const dim3 B(256);
    auto g = [](int n) { return dim3((unsigned)((n + 255) / 256)); };

    // one-time: degrees + f16 copy of x
    k_zero      <<<g(NODES), B, 0, stream>>>(deg, NODES);
    k_deg       <<<g(EDGES), B, 0, stream>>>(dst, deg);
    k_invdeg    <<<g(NODES), B, 0, stream>>>(deg, inv, NODES);
    k_f32_to_f16<<<g(NODES * DIM), B, 0, stream>>>(x, xh, NODES * DIM);

    struct Layer { const float *wl, *wr, *b; const _Float16* in; int Dout, Nh; };
    Layer L[3] = {
        { w1l, w1r, b1, xh, DIM,  128 },
        { w2l, w2r, b2, hA, DIM,  128 },
        { w3l, w3r, b3, hB, NCLS, 48  },   // 40 padded to 48 (3 N-tiles per half)
    };
    _Float16* hout[2] = { hA, hB };

    for (int i = 0; i < 3; ++i) {
        const int Nh = L[i].Nh, N2 = 2 * Nh, NT = N2 / 16, Dout = L[i].Dout;
        const int bpm = (NT + 7) / 8;                       // blocks per 16-row stripe
        // 1) repack [w_l | w_r] -> f16 B fragments
        k_pack<<<g((128 / 32) * NT * 512), B, 0, stream>>>(L[i].wl, L[i].wr, bp, 128, Dout, Nh);
        // 2) Y = h @ [w_l | w_r]   (WMMA; A tile staged in LDS via async copy)
        k_gemm<<<dim3((unsigned)((NODES / 16) * bpm)), B, 0, stream>>>(L[i].in, bp, Yh, NT, bpm);
        // 3) mean-aggregate y_l over edges (linear commuted before aggregation)
        k_zero<<<g(NODES * Nh), B, 0, stream>>>(agg, NODES * Nh);
        k_scatter<<<dim3((unsigned)((EDGES * 32 + 255) / 256)), B, 0, stream>>>(src, dst, Yh, agg, Dout, N2, Nh);
        // 4) relu(agg/deg + y_r + b)
        if (i < 2)
            k_combine16<<<g(NODES * Dout), B, 0, stream>>>(agg, Yh, inv, L[i].b, hout[i], Dout, N2, Nh);
        else
            k_combine32<<<g(NODES * Dout), B, 0, stream>>>(agg, Yh, inv, L[i].b, out, Dout, N2, Nh);
    }
    (void)in_sizes; (void)n_in; (void)out_size; (void)ws_size;
}
